// SparseAttention_87832081203758
// MI455X (gfx1250) — compile-verified, matrix operand order checked by reading, not run on hardware
//
#include <hip/hip_runtime.h>
#include <hip/hip_bf16.h>
#include <math.h>

typedef __bf16 bf16;
typedef __attribute__((ext_vector_type(16))) __bf16 v16bf;
typedef __attribute__((ext_vector_type(8)))  __bf16 v8bf;
typedef __attribute__((ext_vector_type(8)))  float  v8f;
typedef __attribute__((ext_vector_type(4)))  int    v4i;

#define NH 16
#define NKV 4
#define HD 128
#define G_TOK 64
#define S_LEN 2048
#define E_LEN 2112      // S + G
#define EP 2176         // E padded to multiple of 128 for GEMM1
#define HS 2048
#define NQKV 3072       // 16*128 + 4*128 + 4*128
#define TKS 64          // K-step per LDS stage
#define LDT 72          // padded LDS stride (bf16) -> 144B rows, conflict-free

#if defined(__has_builtin)
#if __has_builtin(__builtin_amdgcn_global_load_async_to_lds_b128)
#define ASYNC_LDS 1
#endif
#endif
#ifndef ASYNC_LDS
#define ASYNC_LDS 0
#endif

// ---------------------------------------------------------------------------
// WMMA helpers (CDNA5 16x16x32 bf16, f32 accumulate)
// ---------------------------------------------------------------------------
__device__ inline v16bf make_frag(const bf16* p0) {
  // A/B 16-bit fragment: elements 0..7 at p0[0..7], elements 8..15 at p0[16..23]
  v8bf lo = *(const v8bf*)p0;
  v8bf hi = *(const v8bf*)(p0 + 16);
  v16bf f;
#pragma unroll
  for (int i = 0; i < 8; ++i) { f[i] = lo[i]; f[i + 8] = hi[i]; }
  return f;
}

__device__ inline v8f wmma_bf16(v16bf a, v16bf b, v8f c) {
  return __builtin_amdgcn_wmma_f32_16x16x32_bf16(false, a, false, b, (short)0, c,
                                                 false, false);
}

#if ASYNC_LDS
typedef __attribute__((address_space(1))) v4i as1_v4i;
typedef __attribute__((address_space(3))) v4i as3_v4i;
// 16B global -> LDS async copy (no VGPR round trip, tracked by ASYNCcnt)
__device__ inline void async_copy16(bf16* lds, const bf16* g) {
  as1_v4i* gp = (as1_v4i*)(unsigned long long)(const void*)g;
  as3_v4i* lp = (as3_v4i*)(unsigned)(unsigned long long)(void*)lds;
  __builtin_amdgcn_global_load_async_to_lds_b128(gp, lp, 0, 0);
}
__device__ inline void wait_async_0() {
#if __has_builtin(__builtin_amdgcn_s_wait_asynccnt)
  __builtin_amdgcn_s_wait_asynccnt(0);
#else
  asm volatile("s_wait_asynccnt 0x0" ::: "memory");
#endif
}
// async loads retire in order; <=8 outstanding => previous stage (8 ops) done
__device__ inline void wait_async_8() {
#if __has_builtin(__builtin_amdgcn_s_wait_asynccnt)
  __builtin_amdgcn_s_wait_asynccnt(8);
#else
  asm volatile("s_wait_asynccnt 0x8" ::: "memory");
#endif
}
#endif

// ---------------------------------------------------------------------------
// Prep: x = concat(global_tokens, hidden), f32 -> bf16, zero-pad rows to EP
// ---------------------------------------------------------------------------
__global__ void build_x_kernel(const float* __restrict__ hidden,
                               const float* __restrict__ gtok,
                               bf16* __restrict__ xbf) {
  int e = blockIdx.x;
  int c = blockIdx.y * 256 + threadIdx.x;
  float v = 0.0f;
  if (e < G_TOK)       v = gtok[e * HS + c];
  else if (e < E_LEN)  v = hidden[(e - G_TOK) * HS + c];
  xbf[e * HS + c] = (bf16)v;
}

// WT[n*K + k] = W[k*N + n], f32 -> bf16
__global__ void transpose_w_kernel(const float* __restrict__ W,
                                   bf16* __restrict__ WT, int K, int N) {
  int n = blockIdx.x;
  int k = blockIdx.y * 256 + threadIdx.x;
  WT[n * K + k] = (bf16)W[k * N + n];
}

// ---------------------------------------------------------------------------
// Tiled WMMA GEMM: C[M,N](f32) = A[M,K](bf16) * B[K,N],  B given as BT[N,K]
// block = 256 threads (8 waves), tile 128x128, K-stage 64 (2 WMMA k-steps),
// double-buffered async global->LDS staging (ASYNCcnt pipelined).
// ---------------------------------------------------------------------------
__global__ __launch_bounds__(256) void gemm_bf16_f32(const bf16* __restrict__ A,
                                                     const bf16* __restrict__ BT,
                                                     float* __restrict__ C,
                                                     int M, int N, int K) {
  __shared__ bf16 As[2][128 * LDT];
  __shared__ bf16 Bs[2][128 * LDT];
  const int n0 = blockIdx.x * 128;
  const int m0 = blockIdx.y * 128;
  const int t  = threadIdx.x;
  const int lr = t >> 1;            // 0..127 tile row to load
  const int lc = (t & 1) * 32;      // 0 or 32 (32 cols per thread)
  const int w  = t >> 5;            // wave id 0..7
  const int lane = t & 31;
  const int wr = w >> 1, wc = w & 1;       // 4x2 wave grid -> 32x64 per wave
  const int lm = lane & 15, lh = lane >> 4;

  v8f acc[2][4];
#pragma unroll
  for (int i = 0; i < 2; ++i)
#pragma unroll
    for (int j = 0; j < 4; ++j)
#pragma unroll
      for (int r = 0; r < 8; ++r) acc[i][j][r] = 0.0f;

  const bf16* ga = A  + m0 * K + lr * K + lc;
  const bf16* gb = BT + n0 * K + lr * K + lc;
  const int ldsoff = lr * LDT + lc;

#if ASYNC_LDS
  auto stage_issue = [&](int buf, int k0) {
#pragma unroll
    for (int q = 0; q < 4; ++q) {
      async_copy16(&As[buf][ldsoff + 8 * q], ga + k0 + 8 * q);
      async_copy16(&Bs[buf][ldsoff + 8 * q], gb + k0 + 8 * q);
    }
  };
  stage_issue(0, 0);
#endif

  const int nstage = K / TKS;
  for (int s = 0; s < nstage; ++s) {
    const int k0 = s * TKS;
    const int cur = s & 1;
#if ASYNC_LDS
    if (s + 1 < nstage) {
      stage_issue(cur ^ 1, k0 + TKS);           // overlap next copy with compute
      if (k0 + 2 * TKS < K) {                   // warm L2 two stages ahead
        __builtin_prefetch(ga + k0 + 2 * TKS, 0, 1);
        __builtin_prefetch(gb + k0 + 2 * TKS, 0, 1);
      }
      wait_async_8();                           // stage s landed; s+1 in flight
    } else {
      wait_async_0();
    }
#else
    uint4 ra[4], rb[4];
#pragma unroll
    for (int q = 0; q < 4; ++q) {               // batch loads (no serialization)
      ra[q] = *(const uint4*)(ga + k0 + 8 * q);
      rb[q] = *(const uint4*)(gb + k0 + 8 * q);
    }
#pragma unroll
    for (int q = 0; q < 4; ++q) {
      *(uint4*)&As[cur][ldsoff + 8 * q] = ra[q];
      *(uint4*)&Bs[cur][ldsoff + 8 * q] = rb[q];
    }
#endif
    __syncthreads();
#pragma unroll
    for (int kc = 0; kc < 2; ++kc) {
      v16bf af[2], bfr[4];
#pragma unroll
      for (int i = 0; i < 2; ++i)
        af[i] = make_frag(&As[cur][(wr * 32 + i * 16 + lm) * LDT + kc * 32 + lh * 8]);
#pragma unroll
      for (int j = 0; j < 4; ++j)
        bfr[j] = make_frag(&Bs[cur][(wc * 64 + j * 16 + lm) * LDT + kc * 32 + lh * 8]);
#pragma unroll
      for (int i = 0; i < 2; ++i)
#pragma unroll
        for (int j = 0; j < 4; ++j)
          acc[i][j] = wmma_bf16(af[i], bfr[j], acc[i][j]);
    }
    __syncthreads();     // nobody re-targets this buffer until all have read it
  }

  // C/D layout: VGPR r -> row r + 8*(lane>=16); lane&15 -> column.
  const int crow = m0 + wr * 32 + 8 * lh;
  const int ccol = n0 + wc * 64 + lm;
#pragma unroll
  for (int i = 0; i < 2; ++i)
#pragma unroll
    for (int j = 0; j < 4; ++j)
#pragma unroll
      for (int r = 0; r < 8; ++r)
        C[(crow + i * 16 + r) * N + ccol + j * 16] = acc[i][j][r];
}

// ---------------------------------------------------------------------------
// RoPE + split QKV: Q[NH][E][HD], K[NKV][E][HD] (rope'd), Vt[NKV][HD][E]
// ---------------------------------------------------------------------------
__global__ void rope_split_kernel(const float* __restrict__ qkv,
                                  bf16* __restrict__ Qb, bf16* __restrict__ Kb,
                                  bf16* __restrict__ Vt) {
  int e = blockIdx.x;       // 0..E-1
  int slot = blockIdx.y;    // 0..23 : 16 q heads, 4 k heads, 4 v heads
  int d = threadIdx.x;      // 0..127
  const float* row = qkv + e * NQKV;
  if (slot < NH + NKV) {
    int base = (slot < NH) ? slot * HD : HS + (slot - NH) * HD;
    int dh = d & 63;
    float inv = powf(500000.0f, -((float)(2 * dh)) / 128.0f);
    float ang = (float)e * inv;
    float c = cosf(ang), s = sinf(ang);
    float x = row[base + d];
    float other = (d < 64) ? -row[base + d + 64] : row[base + d - 64];
    bf16 r = (bf16)(x * c + other * s);
    if (slot < NH) Qb[(slot * E_LEN + e) * HD + d] = r;
    else           Kb[((slot - NH) * E_LEN + e) * HD + d] = r;
  } else {
    int h = slot - NH - NKV;
    float x = row[HS + NKV * HD + h * HD + d];
    Vt[(h * HD + d) * E_LEN + e] = (bf16)x;  // transposed for PV WMMA
  }
}

// ---------------------------------------------------------------------------
// Sparse flash attention, transposed formulation.
// One wave per (16-query tile, head). S^T = K*Q^T so query==lane column:
// softmax stats are lane-local (+ one shfl_xor 16), P^T fragment for
// O^T += V^T * P^T is assembled with zero cross-lane traffic.
// ---------------------------------------------------------------------------
__global__ __launch_bounds__(32) void attn_kernel(const bf16* __restrict__ Qb,
                                                  const bf16* __restrict__ Kb,
                                                  const bf16* __restrict__ Vt,
                                                  bf16* __restrict__ Ob) {
  const int qt   = blockIdx.x;           // 0..127  (only output queries e>=G)
  const int head = blockIdx.y;           // 0..15
  const int kvh  = head >> 2;            // GQA: repeat factor 4
  const int q0   = G_TOK + qt * 16;
  const int lane = threadIdx.x & 31;
  const int lm = lane & 15, lh = lane >> 4;
  const float kscale = 0.08838834764831845f * 1.4426950408889634f; // 1/sqrt(HD)*log2(e)

  // Q^T B-fragments: lane -> query q0+lm, contiguous head-dim pairs
  const bf16* qrow = Qb + (head * E_LEN + q0 + lm) * HD;
  v16bf qf[4];
#pragma unroll
  for (int c = 0; c < 4; ++c) qf[c] = make_frag(qrow + c * 32 + lh * 8);

  v8f o[8];
#pragma unroll
  for (int g = 0; g < 8; ++g)
#pragma unroll
    for (int r = 0; r < 8; ++r) o[g][r] = 0.0f;
  float m_run = -1.0e30f, l_run = 0.0f;

  const bf16* kbase = Kb + (kvh * E_LEN + lm) * HD + lh * 8;
  const bf16* vbase0 = Vt + (kvh * HD + lm) * E_LEN + lh * 8;

  const int i_q = q0 + lm;                        // this lane's query index
  const int nkb = (q0 + 16 + 31) >> 5;            // causal block count (32 keys)
  for (int kb = 0; kb < nkb; ++kb) {
    const int c = kb * 32;
    // skip blocks with no strided column and no local-window overlap
    if (((c & 255) != 0) && (c + 31 < q0 - 512)) continue;

    // S^T tiles: keys c..c+15 (sta) and c+16..c+31 (stb)
    v8f sta, stb;
#pragma unroll
    for (int r = 0; r < 8; ++r) { sta[r] = 0.0f; stb[r] = 0.0f; }
    const bf16* krow0 = kbase + c * HD;
    const bf16* krow1 = krow0 + 16 * HD;
#pragma unroll
    for (int ch = 0; ch < 4; ++ch) {
      sta = wmma_bf16(make_frag(krow0 + ch * 32), qf[ch], sta);
      stb = wmma_bf16(make_frag(krow1 + ch * 32), qf[ch], stb);
    }

    // mask (causal AND (local OR strided)), scale into log2 domain
    float pa[8], pb[8];
    float bmax = -3.0e38f;
#pragma unroll
    for (int r = 0; r < 8; ++r) {
      int ja = c + r + 8 * lh;
      int jb = ja + 16;
      float za = sta[r] * kscale;
      float zb = stb[r] * kscale;
      bool oka = (ja <= i_q) && ((ja >= i_q - 512) || ((ja & 255) == 0));
      bool okb = (jb <= i_q) && ((jb >= i_q - 512) || ((jb & 255) == 0));
      pa[r] = oka ? za : -3.0e38f;
      pb[r] = okb ? zb : -3.0e38f;
      bmax = fmaxf(bmax, fmaxf(pa[r], pb[r]));
    }
    bmax = fmaxf(bmax, __shfl_xor(bmax, 16, 32));   // partner lane holds other keys
    float m_new = fmaxf(m_run, bmax);
    float alpha = exp2f(m_run - m_new);

    float ssum = 0.0f;
    v16bf pf;                                        // P^T B-fragment, lane-local
#pragma unroll
    for (int r = 0; r < 8; ++r) {
      float e1 = exp2f(pa[r] - m_new);
      float e2 = exp2f(pb[r] - m_new);
      ssum += e1 + e2;
      pf[r]     = (bf16)e1;
      pf[r + 8] = (bf16)e2;
    }
    ssum += __shfl_xor(ssum, 16, 32);
    l_run = l_run * alpha + ssum;
    m_run = m_new;

#pragma unroll
    for (int g = 0; g < 8; ++g)
#pragma unroll
      for (int r = 0; r < 8; ++r) o[g][r] *= alpha;

    // O^T += V^T * P^T  (A = V^T 16(hd)x32(keys), contiguous from Vt)
#pragma unroll
    for (int g = 0; g < 8; ++g)
      o[g] = wmma_bf16(make_frag(vbase0 + g * 16 * E_LEN + c), pf, o[g]);
  }

  float inv_l = 1.0f / l_run;
  // O^T C-layout: lane column = query lm, rows = head-dim.
  // Each lane owns 8 consecutive elements per group -> one 16B store each.
  bf16* orow = Ob + (q0 - G_TOK + lm) * (NH * HD) + head * HD + 8 * lh;
#pragma unroll
  for (int g = 0; g < 8; ++g) {
    v8bf pk;
#pragma unroll
    for (int r = 0; r < 8; ++r) pk[r] = (bf16)(o[g][r] * inv_l);
    *(v8bf*)(orow + g * 16) = pk;
  }
}

// ---------------------------------------------------------------------------
extern "C" void kernel_launch(void* const* d_in, const int* in_sizes, int n_in,
                              void* d_out, int out_size, void* d_ws, size_t ws_size,
                              hipStream_t stream) {
  (void)in_sizes; (void)n_in; (void)out_size; (void)ws_size;
  const float* hidden = (const float*)d_in[0];
  const float* Wq     = (const float*)d_in[1];
  const float* Wk     = (const float*)d_in[2];
  const float* Wv     = (const float*)d_in[3];
  const float* Wo     = (const float*)d_in[4];
  const float* gtok   = (const float*)d_in[5];
  float* out = (float*)d_out;

  char* ws = (char*)d_ws;
  bf16*  xbf    = (bf16*)ws;  ws += (size_t)EP * HS * sizeof(bf16);        // 8.9 MB
  bf16*  WqkvT  = (bf16*)ws;  ws += (size_t)NQKV * HS * sizeof(bf16);      // 12.6 MB
  bf16*  WoT    = (bf16*)ws;  ws += (size_t)HS * (NH * HD) * sizeof(bf16); // 8.4 MB
  float* qkv    = (float*)ws; ws += (size_t)EP * NQKV * sizeof(float);     // 26.7 MB
  bf16*  Qb     = (bf16*)ws;  ws += (size_t)NH * E_LEN * HD * sizeof(bf16);
  bf16*  Kb     = (bf16*)ws;  ws += (size_t)NKV * E_LEN * HD * sizeof(bf16);
  bf16*  Vt     = (bf16*)ws;  ws += (size_t)NKV * HD * E_LEN * sizeof(bf16);
  bf16*  attnO  = (bf16*)ws;  ws += (size_t)S_LEN * (NH * HD) * sizeof(bf16);

  // 1) prep
  build_x_kernel<<<dim3(EP, HS / 256), 256, 0, stream>>>(hidden, gtok, xbf);
  transpose_w_kernel<<<dim3(2048, HS / 256), 256, 0, stream>>>(Wq, WqkvT, HS, 2048);
  transpose_w_kernel<<<dim3(512,  HS / 256), 256, 0, stream>>>(Wk, WqkvT + (size_t)2048 * HS, HS, 512);
  transpose_w_kernel<<<dim3(512,  HS / 256), 256, 0, stream>>>(Wv, WqkvT + (size_t)2560 * HS, HS, 512);
  transpose_w_kernel<<<dim3(2048, (NH * HD) / 256), 256, 0, stream>>>(Wo, WoT, NH * HD, 2048);

  // 2) QKV = x @ Wqkv   (M=2176, N=3072, K=2048)
  gemm_bf16_f32<<<dim3(NQKV / 128, EP / 128), 256, 0, stream>>>(xbf, WqkvT, qkv, EP, NQKV, HS);

  // 3) RoPE + split
  rope_split_kernel<<<dim3(E_LEN, NH + 2 * NKV), 128, 0, stream>>>(qkv, Qb, Kb, Vt);

  // 4) sparse flash attention (only queries e >= G contribute to output)
  attn_kernel<<<dim3(S_LEN / 16, NH), 32, 0, stream>>>(Qb, Kb, Vt, attnO);

  // 5) out = attnO @ Wo  (M=2048, N=2048, K=2048)
  gemm_bf16_f32<<<dim3(HS / 128, S_LEN / 128), 256, 0, stream>>>(attnO, WoT, out, S_LEN, HS, NH * HD);
}
